// vis_net_19353122635821
// MI455X (gfx1250) — matrix operand order repaired and checked by reading, not compiled, not run on hardware
//
#include <hip/hip_runtime.h>

typedef __attribute__((ext_vector_type(16))) _Float16 v16h;
typedef __attribute__((ext_vector_type(8)))  float    v8f;

#define LVLS 8
#define LOG2T 19
#define TSZ (1u << LOG2T)

// DPP-based add with compile-time control (reduction within 16-lane rows).
template <int CTRL>
__device__ __forceinline__ float dpp_add(float v) {
    int d = __builtin_amdgcn_update_dpp(0, __float_as_int(v), CTRL, 0xF, 0xF, true);
    return v + __int_as_float(d);
}

// ds_swizzle xor-16 (group-of-32 mode: xor=0x10, or=0, and=0x1f)
__device__ __forceinline__ float swz_xor16(float v) {
    return __int_as_float(__builtin_amdgcn_ds_swizzle(__float_as_int(v), 0x401F));
}

__launch_bounds__(256)
__global__ void visnet_kernel(const float* __restrict__ x,
                              const float* __restrict__ table,
                              const float* __restrict__ W1,
                              const float* __restrict__ b1,
                              const float* __restrict__ W2,
                              const float* __restrict__ b2,
                              float* __restrict__ out, int N)
{
    // 8 waves * 32 points * 8 packed-f16-pair words = 8 KB
    __shared__ __align__(16) unsigned int lds_enc[8][32][8];

    const int lane = threadIdx.x & 31;
    const int wave = threadIdx.x >> 5;
    const int col  = lane & 15;
    const int half = lane >> 4;

    // ---- B fragments: W1 (16x32 f32 row-major) as f16, K padded 16->32 ----
    // B layout (32x16 f16): lanes 0-15 hold K=0..7 (elems 0-7) for col N=lane&15,
    // lanes 16-31 hold K=8..15 (elems 0-7); elems 8-15 are K>=16 (zero padding).
    v16h bf0, bf1;
#pragma unroll
    for (int k = 0; k < 16; ++k) { bf0[k] = (_Float16)0.0f; bf1[k] = (_Float16)0.0f; }
    {
#pragma unroll
        for (int k = 0; k < 8; ++k) {
            bf0[k] = (_Float16)W1[(half * 8 + k) * 32 + col];
            bf1[k] = (_Float16)W1[(half * 8 + k) * 32 + col + 16];
        }
    }
    const float bn0 = b1[col];
    const float bn1 = b1[col + 16];
    const float w2a = W2[col];
    const float w2b = W2[col + 16];
    const float b2s = b2[0];

    // scale_l = 32 * (64^(l/7)) - 1
    const float SC[LVLS] = {31.0f, 56.9663145f, 104.0029256f, 189.2072672f,
                            343.5504490f, 623.1349820f, 1129.5876860f, 2047.0f};

    const int waveGlobal = blockIdx.x * 8 + wave;
    const int nWaves     = gridDim.x * 8;
    const int nChunks    = (N + 31) >> 5;

    for (int ch = waveGlobal; ch < nChunks; ch += nWaves) {
        const int base = ch * 32;
        int i = base + lane;
        if (i >= N) i = N - 1;             // clamp: keep EXEC all-ones for WMMA
        const float x0 = (x[i * 3 + 0] + 1.0f) * 0.5f;
        const float x1 = (x[i * 3 + 1] + 1.0f) * 0.5f;
        const float x2 = (x[i * 3 + 2] + 1.0f) * 0.5f;

        // ---- hash-grid encoding: one point per lane ----
        // Fully unrolled: packed[] stays in statically-indexed registers
        // (no movrel) and all 64 independent b64 gathers can be in flight.
        unsigned int packed[LVLS];
#pragma unroll
        for (int l = 0; l < LVLS; ++l) {
            const float s = SC[l];
            float px = fmaf(x0, s, 0.5f);
            float py = fmaf(x1, s, 0.5f);
            float pz = fmaf(x2, s, 0.5f);
            float fx = floorf(px), fy = floorf(py), fz = floorf(pz);
            float wx = px - fx, wy = py - fy, wz = pz - fz;
            unsigned int ux = (unsigned int)fx;
            unsigned int uy = (unsigned int)fy;
            unsigned int uz = (unsigned int)fz;

            // strength-reduced hash components
            unsigned int hx0 = ux,               hx1 = ux + 1u;
            unsigned int hy0 = uy * 2654435761u, hy1 = hy0 + 2654435761u;
            unsigned int hz0 = uz * 805459861u,  hz1 = hz0 + 805459861u;

            // trilinear weight factors
            float ox = 1.0f - wx, oy = 1.0f - wy, oz = 1.0f - wz;
            float w00 = ox * oy, w10 = wx * oy, w01 = ox * wy, w11 = wx * wy;

            const float* tl = table + (size_t)l * (size_t)(TSZ * 2u);
            float a0 = 0.0f, a1 = 0.0f;
#pragma unroll
            for (int c = 0; c < 8; ++c) {
                unsigned int h = ((c & 1) ? hx1 : hx0)
                               ^ (((c >> 1) & 1) ? hy1 : hy0)
                               ^ ((c >> 2) ? hz1 : hz0);
                unsigned int idx = h & (TSZ - 1u);
                float2 f = *(const float2*)(tl + (size_t)idx * 2u);  // b64 gather
                float wxy = ((c & 1) ? ((c >> 1) & 1 ? w11 : w10)
                                     : ((c >> 1) & 1 ? w01 : w00));
                float wt = wxy * ((c >> 2) ? wz : oz);
                a0 = fmaf(f.x, wt, a0);
                a1 = fmaf(f.y, wt, a1);
            }
            union { _Float16 h2[2]; unsigned int u; } pk;
            pk.h2[0] = (_Float16)a0;
            pk.h2[1] = (_Float16)a1;
            packed[l] = pk.u;              // feature pair (2l, 2l+1)
        }

        *(uint4*)&lds_enc[wave][lane][0] = make_uint4(packed[0], packed[1], packed[2], packed[3]);
        *(uint4*)&lds_enc[wave][lane][4] = make_uint4(packed[4], packed[5], packed[6], packed[7]);

        // same-wave producer->consumer via LDS: DS ops are in-order per wave
        asm volatile("s_wait_dscnt 0x0" ::: "memory");

        // ---- two 16-point tiles: layer1 WMMA, layer2 DPP reduction ----
        float tileVal[2];                  // per-tile output, valid in lanes 0-15
#pragma unroll
        for (int t = 0; t < 2; ++t) {
            // A layout (16x32 f16): lanes 0-15 row M=col hold K=0..7,
            // lanes 16-31 row M=col hold K=8..15; upper K zero-padded.
            union { uint4 u4[2]; v16h h; } af;
            af.u4[0] = *(const uint4*)&lds_enc[wave][t * 16 + col][half * 4];
            af.u4[1] = make_uint4(0u, 0u, 0u, 0u);

            v8f c0, c1;
#pragma unroll
            for (int r = 0; r < 8; ++r) { c0[r] = bn0; c1[r] = bn1; }  // bias preload

            c0 = __builtin_amdgcn_wmma_f32_16x16x32_f16(false, af.h, false, bf0,
                                                        (short)0, c0, false, false);
            c1 = __builtin_amdgcn_wmma_f32_16x16x32_f16(false, af.h, false, bf1,
                                                        (short)0, c1, false, false);

            // per-lane partial: relu + dot with W2 column slice
            float v[8];
#pragma unroll
            for (int r = 0; r < 8; ++r) {
                float h0 = fmaxf(c0[r], 0.0f);
                float h1 = fmaxf(c1[r], 0.0f);
                v[r] = fmaf(h0, w2a, h1 * w2b);
                // butterfly over each 16-lane half; xor masks {1,2,7,15}
                v[r] = dpp_add<0xB1>(v[r]);   // quad_perm [1,0,3,2]  (xor 1)
                v[r] = dpp_add<0x4E>(v[r]);   // quad_perm [2,3,0,1]  (xor 2)
                v[r] = dpp_add<0x141>(v[r]);  // row_half_mirror      (xor 7)
                v[r] = dpp_add<0x140>(v[r]);  // row_mirror           (xor 15)
            }
            // select v[lane&7] via cndmask tree
            float a0s = (lane & 1) ? v[1] : v[0];
            float a1s = (lane & 1) ? v[3] : v[2];
            float a2s = (lane & 1) ? v[5] : v[4];
            float a3s = (lane & 1) ? v[7] : v[6];
            float b0s = (lane & 2) ? a1s : a0s;
            float b1s = (lane & 2) ? a3s : a2s;
            float cs  = (lane & 4) ? b1s : b0s;
            // rows 8..15 live in the upper half: pull across with xor-16 swizzle
            float ds  = (lane & 8) ? swz_xor16(cs) : cs;
            tileVal[t] = ds;               // lane m (0-15) = row m of tile t
        }

        // merge tiles: lanes 0-15 <- tile0 rows, lanes 16-31 <- tile1 rows
        float merged = (lane & 16) ? swz_xor16(tileVal[1]) : tileVal[0];

        // sigmoid (fast rcp) + single coalesced store of 32 outputs
        float z = merged + b2s;
        float e = __expf(-z);
        float sig = __builtin_amdgcn_rcpf(1.0f + e);
        if (base + lane < N) out[base + lane] = sig;
    }
}

extern "C" void kernel_launch(void* const* d_in, const int* in_sizes, int n_in,
                              void* d_out, int out_size, void* d_ws, size_t ws_size,
                              hipStream_t stream) {
    const float* x   = (const float*)d_in[0];
    const float* tbl = (const float*)d_in[1];
    const float* W1  = (const float*)d_in[2];
    const float* b1  = (const float*)d_in[3];
    const float* W2  = (const float*)d_in[4];
    const float* b2  = (const float*)d_in[5];
    float* out = (float*)d_out;

    const int N = in_sizes[0] / 3;
    const int nChunks = (N + 31) / 32;           // 32 points per wave-chunk
    int blocks = (nChunks + 7) / 8;              // 8 waves per block
    if (blocks > 4096) blocks = 4096;            // persistent grid-stride
    visnet_kernel<<<blocks, 256, 0, stream>>>(x, tbl, W1, b1, W2, b2, out, N);
}
